// AiLUT_6871947673791
// MI455X (gfx1250) — matrix operand order repaired and checked by reading, not compiled, not run on hardware
//
#include <hip/hip_runtime.h>

// ---------------- constants -------------------------------------------------
#define DD    33
#define D3    35937            // 33^3
#define NFEAT 512
#define HWF   (1024*1024)      // full-res H*W
#define BATCH 4
#define OUT_PIX (BATCH*3*HWF)  // 12582912

typedef __attribute__((ext_vector_type(16))) _Float16 v16h;
typedef __attribute__((ext_vector_type(8)))  float    v8f;
typedef float vf4 __attribute__((ext_vector_type(4)));

struct f3 { float x, y, z; };   // 12B, 4B-aligned: one trilinear corner, 3 channels

__device__ __forceinline__ int imin(int a, int b) { return a < b ? a : b; }
__device__ __forceinline__ int imax(int a, int b) { return a > b ? a : b; }

// ---------------- 1. bilinear resize 1024 -> 256 ---------------------------
__global__ void k_resize(const float* __restrict__ lq, float* __restrict__ x0) {
  int t = blockIdx.x * blockDim.x + threadIdx.x;     // over 4*3*256*256
  if (t >= BATCH * 3 * 256 * 256) return;
  int j  = t & 255;
  int i  = (t >> 8) & 255;
  int bc = t >> 16;
  // jax bilinear (antialias=False): src = (i+0.5)*4 - 0.5 = 4i + 1.5
  float sy = 4.f * i + 1.5f, sx = 4.f * j + 1.5f;
  int y0 = (int)sy, x0i = (int)sx;                   // 4i+1 in [1,1021]
  float fy = sy - (float)y0, fx = sx - (float)x0i;   // 0.5
  const float* p = lq + (size_t)bc * HWF;
  float v00 = p[y0 * 1024 + x0i];
  float v01 = p[y0 * 1024 + x0i + 1];
  float v10 = p[(y0 + 1) * 1024 + x0i];
  float v11 = p[(y0 + 1) * 1024 + x0i + 1];
  x0[t] = (v00 * (1.f - fx) + v01 * fx) * (1.f - fy)
        + (v10 * (1.f - fx) + v11 * fx) * fy;
}

// ---------------- 2. stride-2 3x3 conv as implicit GEMM via WMMA -----------
// One wave computes a 16-pixel x 16-out-channel tile, f16 A/B, f32 accumulate.
// Fragment striping per CDNA5 ISA 7.12.2 (wave32).
template <int CI, int CO, int HI, int WI>
__global__ __launch_bounds__(32) void k_conv_wmma(const float* __restrict__ x,
                                                  const float* __restrict__ w,
                                                  const float* __restrict__ bias,
                                                  float* __restrict__ y) {
  constexpr int HO = HI / 2, WO = WI / 2;
  constexpr int KT = CI * 9;
  const int lane = threadIdx.x;
  const int pt   = blockIdx.x * 16;        // pixel tile base (within batch)
  const int co0  = blockIdx.y * 16;        // out-channel tile base
  const int b    = blockIdx.z;
  const int half = (lane >= 16) ? 1 : 0;
  const int ln   = lane & 15;

  const int p  = pt + ln;                  // this lane's A row (pixel)
  const int ho = p / WO, wo = p % WO;
  const int co = co0 + ln;                 // this lane's B column (out channel)

  v8f c = {};
  for (int k0 = 0; k0 < KT; k0 += 32) {
    v16h a, bm;
#pragma unroll
    for (int e = 0; e < 16; ++e) {
      const int v = e >> 1, h = e & 1;
      // ---- A element (im2col gather) ----
      int ka = k0 + ((v >> 2) * 16) + half * 8 + ((v & 3) * 2) + h;
      float av = 0.f;
      if (ka < KT) {
        int ci = ka / 9, rem = ka % 9, ky = rem / 3, kx = rem % 3;
        int hi = ho * 2 - 1 + ky, wi = wo * 2 - 1 + kx;   // stride 2, pad 1
        if (hi >= 0 && hi < HI && wi >= 0 && wi < WI)
          av = x[(((size_t)b * CI + ci) * HI + hi) * WI + wi];
      }
      a[e] = (_Float16)av;
      // ---- B element (weight gather) ----
      int kb = k0 + ((v >> 2) * 16) + half * 8 + ((v & 3) * 2) + h;
      float bv = 0.f;
      if (kb < KT) {
        int ci = kb / 9, rem = kb % 9, ky = rem / 3, kx = rem % 3;
        bv = w[((co * CI + ci) * 3 + ky) * 3 + kx];
      }
      bm[e] = (_Float16)bv;
    }
    c = __builtin_amdgcn_wmma_f32_16x16x32_f16(false, a, false, bm,
                                               (short)0, c, false, false);
  }
  // epilogue: bias + leaky_relu(0.2), NCHW store
#pragma unroll
  for (int r = 0; r < 8; ++r) {
    int m  = r + half * 8;
    int pp = pt + m;
    int oh = pp / WO, ow = pp % WO;
    float val = c[r] + bias[co];
    val = (val >= 0.f) ? val : 0.2f * val;
    y[(((size_t)b * CO + co) * HO + oh) * WO + ow] = val;
  }
}

// ---------------- 3. instance norm (two-pass, float4) ----------------------
__global__ void k_inorm_stats(const float* __restrict__ x, float* __restrict__ mv,
                              int HW) {
  const int bc = blockIdx.x;
  const vf4* p4 = (const vf4*)(x + (size_t)bc * HW);
  const int n4 = HW >> 2;
  float s = 0.f, s2 = 0.f;
  for (int i = threadIdx.x; i < n4; i += blockDim.x) {
    vf4 v = p4[i];
    s  += v.x + v.y + v.z + v.w;
    s2 += v.x * v.x + v.y * v.y + v.z * v.z + v.w * v.w;
  }
  for (int off = 16; off > 0; off >>= 1) {
    s  += __shfl_xor(s, off);
    s2 += __shfl_xor(s2, off);
  }
  __shared__ float sa[8], sb[8];
  int wid = threadIdx.x >> 5;
  if ((threadIdx.x & 31) == 0) { sa[wid] = s; sb[wid] = s2; }
  __syncthreads();
  if (threadIdx.x == 0) {
    float S = 0.f, S2 = 0.f;
    for (int i = 0; i < (int)(blockDim.x >> 5); ++i) { S += sa[i]; S2 += sb[i]; }
    float m = S / (float)HW;
    mv[2 * bc]     = m;
    mv[2 * bc + 1] = S2 / (float)HW - m * m;
  }
}

__global__ void k_inorm_apply(float* __restrict__ x, const float* __restrict__ mv,
                              const float* __restrict__ g, const float* __restrict__ be,
                              int C, int HW4, int total4) {
  int t = blockIdx.x * blockDim.x + threadIdx.x;   // float4 index
  if (t >= total4) return;
  int bc = t / HW4;
  int c  = bc % C;
  float m = mv[2 * bc], v = mv[2 * bc + 1];
  float sc = rsqrtf(v + 1e-5f) * g[c];
  float sh = be[c] - m * sc;
  vf4* x4 = (vf4*)x;
  vf4 q = x4[t];
  q.x = q.x * sc + sh; q.y = q.y * sc + sh;
  q.z = q.z * sc + sh; q.w = q.w * sc + sh;
  x4[t] = q;
}

// ---------------- 4. 4x4 mean pool (B,128,8,8)->(B,512) codes --------------
__global__ void k_pool(const float* __restrict__ x5, float* __restrict__ codes) {
  int t = blockIdx.x * blockDim.x + threadIdx.x;
  if (t >= BATCH * 128 * 4) return;
  int b = t >> 9, rem = t & 511;
  int c = rem >> 2, q = rem & 3;
  int py = q >> 1, px = q & 1;
  const float* p = x5 + (size_t)(b * 128 + c) * 64;
  float s = 0.f;
#pragma unroll
  for (int u = 0; u < 4; ++u)
#pragma unroll
    for (int v = 0; v < 4; ++v) s += p[(py * 4 + u) * 8 + (px * 4 + v)];
  codes[b * NFEAT + c * 4 + py * 2 + px] = s * (1.f / 16.f);
}

// ---------------- 5. FC heads: weights(B,3) and intervals(B,96) ------------
__global__ void k_fc(const float* __restrict__ codes,
                     const float* __restrict__ wg_w, const float* __restrict__ lut_wd,
                     const float* __restrict__ wg_b,
                     const float* __restrict__ ai_w, const float* __restrict__ ai_wd,
                     const float* __restrict__ ai_b,
                     float* __restrict__ wts_ws, float* __restrict__ wts_out,
                     float* __restrict__ intervals) {
  int t = blockIdx.x * blockDim.x + threadIdx.x;
  if (t >= BATCH * 99) return;
  int b = t / 99, j = t % 99;
  const float* cd = codes + b * NFEAT;
  float acc = 0.f;
  if (j < 3) {
    for (int k = 0; k < NFEAT; ++k)
      acc += cd[k] * wg_w[j * NFEAT + k] * (1.f + lut_wd[j * NFEAT + k]);
    acc += wg_b[j];
    wts_ws[b * 3 + j]  = acc;
    wts_out[b * 3 + j] = acc;
  } else {
    int jj = j - 3;
    for (int k = 0; k < NFEAT; ++k)
      acc += cd[k] * ai_w[jj * NFEAT + k] * (1.f + ai_wd[jj * NFEAT + k]);
    acc += ai_b[jj];
    intervals[b * 96 + jj] = acc;
  }
}

// ---------------- 6. softmax + cumsum -> vertices (B,3,33) -----------------
__global__ void k_softmax_vert(const float* __restrict__ intervals,
                               float* __restrict__ verts_ws,
                               float* __restrict__ verts_out) {
  int t = threadIdx.x;
  if (t >= BATCH * 3) return;
  int b = t / 3, c = t % 3;
  const float* row = intervals + b * 96 + c * 32;
  float mx = row[0];
#pragma unroll
  for (int i = 1; i < 32; ++i) mx = fmaxf(mx, row[i]);
  float e[32];
  float s = 0.f;
#pragma unroll
  for (int i = 0; i < 32; ++i) { e[i] = expf(row[i] - mx); s += e[i]; }
  float inv = 1.f / s, cum = 0.f;
  int base = b * 99 + c * 33;
  verts_ws[base]  = 0.f;
  verts_out[base] = 0.f;
#pragma unroll
  for (int i = 0; i < 32; ++i) {
    cum += e[i] * inv;
    verts_ws[base + 1 + i]  = cum;
    verts_out[base + 1 + i] = cum;
  }
}

// ---------------- 7. materialize LUTs, channel-INTERLEAVED (b, lin, c) -----
// lutsI[b][lin][c] so that one trilinear corner = one contiguous 12B triple.
__global__ void k_luts(const float* __restrict__ wts, const float* __restrict__ bank,
                       float* __restrict__ lutsI) {
  int t = blockIdx.x * blockDim.x + threadIdx.x;     // over BATCH*D3
  if (t >= BATCH * D3) return;
  int b = t / D3, lin = t % D3;
  float w0 = wts[b * 3], w1 = wts[b * 3 + 1], w2 = wts[b * 3 + 2];
  float* o = lutsI + ((size_t)b * D3 + lin) * 3;
#pragma unroll
  for (int c = 0; c < 3; ++c) {
    const float* br = bank + (size_t)(c * D3 + lin) * 3;
    o[c] = w0 * br[0] + w1 * br[1] + w2 * br[2];
  }
}

// ---------------- 8. per-pixel AiLUT transform (HBM-bound pass) ------------
// branchless 6-step binary search over 33 LDS-resident vertices,
// searchsorted(side='right') semantics, idx clipped to [1,32].
__device__ __forceinline__ float ailut_coord(float q, const float* __restrict__ v) {
  int idx = 0;                              // last index with v[idx] <= q
  if (v[idx + 16] <= q) idx += 16;
  if (v[idx + 8]  <= q) idx += 8;
  if (v[idx + 4]  <= q) idx += 4;
  if (v[idx + 2]  <= q) idx += 2;
  if (idx + 1 <= 32 && v[idx + 1] <= q) idx += 1;
  if (idx + 1 <= 32 && v[idx + 1] <= q) idx += 1;
  int ss = imin(imax(idx + 1, 1), 32);      // clip(searchsorted, 1, 32)
  float vlo = v[ss - 1], vhi = v[ss];
  float frac = (q - vlo) / (vhi - vlo + 1e-8f);
  float coord = (float)(ss - 1) + frac;
  return fminf(fmaxf(coord, 0.f), 32.f);
}

__device__ __forceinline__ void ailut_pixel(float q0, float q1, float q2,
                                            const float* __restrict__ sv,
                                            const float* __restrict__ LB,
                                            float& o0, float& o1, float& o2) {
  float gx = ailut_coord(q0, sv);
  float gy = ailut_coord(q1, sv + 33);
  float gz = ailut_coord(q2, sv + 66);
  float xf = floorf(gx), yf = floorf(gy), zf = floorf(gz);
  float wx = gx - xf, wy = gy - yf, wz = gz - zf;
  int x0 = imin(imax((int)xf, 0), 32);
  int y0 = imin(imax((int)yf, 0), 32);
  int z0 = imin(imax((int)zf, 0), 32);
  int x1 = imin(x0 + 1, 32), y1 = imin(y0 + 1, 32), z1 = imin(z0 + 1, 32);
  // 8 corners; each load returns all 3 channels (interleaved layout)
  const f3 c000 = *(const f3*)(LB + (size_t)((z0 * DD + y0) * DD + x0) * 3);
  const f3 c001 = *(const f3*)(LB + (size_t)((z0 * DD + y0) * DD + x1) * 3);
  const f3 c010 = *(const f3*)(LB + (size_t)((z0 * DD + y1) * DD + x0) * 3);
  const f3 c011 = *(const f3*)(LB + (size_t)((z0 * DD + y1) * DD + x1) * 3);
  const f3 c100 = *(const f3*)(LB + (size_t)((z1 * DD + y0) * DD + x0) * 3);
  const f3 c101 = *(const f3*)(LB + (size_t)((z1 * DD + y0) * DD + x1) * 3);
  const f3 c110 = *(const f3*)(LB + (size_t)((z1 * DD + y1) * DD + x0) * 3);
  const f3 c111 = *(const f3*)(LB + (size_t)((z1 * DD + y1) * DD + x1) * 3);
#define TRI_(f)                                                              \
  (((c000.f * (1.f - wx) + c001.f * wx) * (1.f - wy) +                       \
    (c010.f * (1.f - wx) + c011.f * wx) * wy) * (1.f - wz) +                 \
   ((c100.f * (1.f - wx) + c101.f * wx) * (1.f - wy) +                       \
    (c110.f * (1.f - wx) + c111.f * wx) * wy) * wz)
  o0 = fminf(fmaxf(TRI_(x), 0.f), 1.f);
  o1 = fminf(fmaxf(TRI_(y), 0.f), 1.f);
  o2 = fminf(fmaxf(TRI_(z), 0.f), 1.f);
#undef TRI_
}

__global__ void k_ailut(const float* __restrict__ lq, const float* __restrict__ lutsI,
                        const float* __restrict__ verts, float* __restrict__ out) {
  __shared__ float sv[99];
  const int b = blockIdx.y;
  if (threadIdx.x < 99) sv[threadIdx.x] = verts[b * 99 + threadIdx.x];
  __syncthreads();
  const int nq = HWF / 4;                               // float4 count per channel
  int i = blockIdx.x * blockDim.x + threadIdx.x;
  if (i >= nq) return;
  const vf4* lq4 = (const vf4*)(lq + (size_t)b * 3 * HWF);
  vf4* out4      = (vf4*)(out + (size_t)b * 3 * HWF);
  // streaming traffic: non-temporal so L2 stays reserved for the LUT gathers
  vf4 q0 = __builtin_nontemporal_load(lq4 + i);
  vf4 q1 = __builtin_nontemporal_load(lq4 + nq + i);
  vf4 q2 = __builtin_nontemporal_load(lq4 + 2 * nq + i);
  const float* LB = lutsI + (size_t)b * 3 * D3;
  vf4 r0, r1, r2;
  {
    float a, bb, c;
    ailut_pixel(q0.x, q1.x, q2.x, sv, LB, a, bb, c); r0.x = a; r1.x = bb; r2.x = c;
    ailut_pixel(q0.y, q1.y, q2.y, sv, LB, a, bb, c); r0.y = a; r1.y = bb; r2.y = c;
    ailut_pixel(q0.z, q1.z, q2.z, sv, LB, a, bb, c); r0.z = a; r1.z = bb; r2.z = c;
    ailut_pixel(q0.w, q1.w, q2.w, sv, LB, a, bb, c); r0.w = a; r1.w = bb; r2.w = c;
  }
  __builtin_nontemporal_store(r0, out4 + i);
  __builtin_nontemporal_store(r1, out4 + nq + i);
  __builtin_nontemporal_store(r2, out4 + 2 * nq + i);
}

// ---------------- launch ----------------------------------------------------
extern "C" void kernel_launch(void* const* d_in, const int* in_sizes, int n_in,
                              void* d_out, int out_size, void* d_ws, size_t ws_size,
                              hipStream_t stream) {
  const float* lq     = (const float*)d_in[0];
  const float* lut_wd = (const float*)d_in[1];
  const float* ai_wd  = (const float*)d_in[2];
  const float* cw[5]  = {(const float*)d_in[3], (const float*)d_in[5],
                         (const float*)d_in[7], (const float*)d_in[9],
                         (const float*)d_in[11]};
  const float* cb[5]  = {(const float*)d_in[4], (const float*)d_in[6],
                         (const float*)d_in[8], (const float*)d_in[10],
                         (const float*)d_in[12]};
  const float* g[4]   = {(const float*)d_in[13], (const float*)d_in[15],
                         (const float*)d_in[17], (const float*)d_in[19]};
  const float* be[4]  = {(const float*)d_in[14], (const float*)d_in[16],
                         (const float*)d_in[18], (const float*)d_in[20]};
  const float* wg_w   = (const float*)d_in[21];
  const float* wg_b   = (const float*)d_in[22];
  const float* bank   = (const float*)d_in[23];
  const float* ai_w   = (const float*)d_in[24];
  const float* ai_b   = (const float*)d_in[25];
  float* out = (float*)d_out;

  // workspace carve-up (floats)
  float* ws        = (float*)d_ws;
  float* x0        = ws;                    // 4*3*256*256   = 786432
  float* x1        = x0 + 786432;           // 4*16*128*128  = 1048576
  float* x2        = x1 + 1048576;          // 4*32*64*64    = 524288
  float* x3        = x2 + 524288;           // 4*64*32*32    = 262144
  float* x4        = x3 + 262144;           // 4*128*16*16   = 131072
  float* x5        = x4 + 131072;           // 4*128*8*8     = 32768
  float* codes     = x5 + 32768;            // 4*512         = 2048
  float* stats     = codes + 2048;          // up to 4*128*2 = 1024
  float* intervals = stats + 1024;          // 4*96          = 384
  float* verts     = intervals + 384;       // 4*99          = 396
  float* wts       = verts + 396;           // 4*3           = 12
  float* lutsI     = wts + 12;              // 4*35937*3     = 431244 (interleaved)

  // 1. resize
  k_resize<<<(BATCH * 3 * 256 * 256 + 255) / 256, 256, 0, stream>>>(lq, x0);

  // 2-6. backbone: conv(+lrelu) [+ instance norm]
  k_conv_wmma<3, 16, 256, 256><<<dim3(1024, 1, BATCH), 32, 0, stream>>>(x0, cw[0], cb[0], x1);
  k_inorm_stats<<<BATCH * 16, 256, 0, stream>>>(x1, stats, 128 * 128);
  k_inorm_apply<<<(262144 + 255) / 256, 256, 0, stream>>>(x1, stats, g[0], be[0], 16, 4096, 262144);

  k_conv_wmma<16, 32, 128, 128><<<dim3(256, 2, BATCH), 32, 0, stream>>>(x1, cw[1], cb[1], x2);
  k_inorm_stats<<<BATCH * 32, 256, 0, stream>>>(x2, stats, 64 * 64);
  k_inorm_apply<<<(131072 + 255) / 256, 256, 0, stream>>>(x2, stats, g[1], be[1], 32, 1024, 131072);

  k_conv_wmma<32, 64, 64, 64><<<dim3(64, 4, BATCH), 32, 0, stream>>>(x2, cw[2], cb[2], x3);
  k_inorm_stats<<<BATCH * 64, 256, 0, stream>>>(x3, stats, 32 * 32);
  k_inorm_apply<<<(65536 + 255) / 256, 256, 0, stream>>>(x3, stats, g[2], be[2], 64, 256, 65536);

  k_conv_wmma<64, 128, 32, 32><<<dim3(16, 8, BATCH), 32, 0, stream>>>(x3, cw[3], cb[3], x4);
  k_inorm_stats<<<BATCH * 128, 256, 0, stream>>>(x4, stats, 16 * 16);
  k_inorm_apply<<<(32768 + 255) / 256, 256, 0, stream>>>(x4, stats, g[3], be[3], 128, 64, 32768);

  k_conv_wmma<128, 128, 16, 16><<<dim3(4, 8, BATCH), 32, 0, stream>>>(x4, cw[4], cb[4], x5);

  // 7. pool -> codes
  k_pool<<<(BATCH * 128 * 4 + 255) / 256, 256, 0, stream>>>(x5, codes);

  // 8. FC heads (weights + intervals); weights also go to d_out tail
  k_fc<<<(BATCH * 99 + 255) / 256, 256, 0, stream>>>(
      codes, wg_w, lut_wd, wg_b, ai_w, ai_wd, ai_b,
      wts, out + OUT_PIX, intervals);

  // 9. softmax + cumsum -> vertices (ws copy + d_out tail)
  k_softmax_vert<<<1, 32, 0, stream>>>(intervals, verts, out + OUT_PIX + BATCH * 3);

  // 10. materialize per-batch interleaved LUTs
  k_luts<<<(BATCH * D3 + 255) / 256, 256, 0, stream>>>(wts, bank, lutsI);

  // 11. the memory-bound pass: per-pixel adaptive-interval trilinear LUT
  k_ailut<<<dim3((HWF / 4 + 255) / 256, BATCH), 256, 0, stream>>>(lq, lutsI, verts, out);
}
